// NavieUNet_V1_68410239091227
// MI455X (gfx1250) — matrix operand order repaired and checked by reading, not compiled, not run on hardware
//
#include <hip/hip_runtime.h>
#include <hip/hip_bf16.h>

// ---------------------------------------------------------------------------
// GCN U-Net forward for MI455X (gfx1250, wave32).
// Dense GEMMs -> v_wmma_f32_16x16x32_bf16 with pre-packed bf16 fragments:
//   inner loop = 2x 32B vector loads + 1 WMMA, no guards, no branches.
// Graph aggregation -> native f32 global atomics.
// ---------------------------------------------------------------------------

typedef __attribute__((ext_vector_type(16))) __bf16 v16bf;
typedef __attribute__((ext_vector_type(8)))  float  v8f;

#define N0_ 16384
#define N1_ 4096
#define N2_ 1024
#define E0_ (N0_ * 16)
#define E1_ (N1_ * 16)
#define E2_ (N2_ * 16)

// ---------------------------------------------------------------------------
// Fragment packing (contiguous k order per lane half), fp32 -> bf16, zero-padded to K32.
// A fragment (16x32, M-major): Ap[((kt*M + m)*2 + h)*16 + j],
//   k = kt*32 + 16h + j
// B fragment (32x16, N-col per lane): Wp[((kt*N + n)*2 + h)*16 + j],
//   k = kt*32 + 16h + j
// ---------------------------------------------------------------------------
__global__ void pack_a_kernel(const float* __restrict__ A, __bf16* __restrict__ Ap,
                              int M, int K, int K32) {
    size_t t = (size_t)blockIdx.x * blockDim.x + threadIdx.x;
    if (t >= (size_t)M * K32) return;
    int j = (int)(t & 15);
    int h = (int)((t >> 4) & 1);
    size_t r = t >> 5;
    int m  = (int)(r % M);
    int kt = (int)(r / M);
    int v = j >> 1, p = j & 1;
    int k = kt * 32 + 16 * h + j;
    Ap[t] = (k < K) ? (__bf16)A[(size_t)m * K + k] : (__bf16)0.0f;
}

__global__ void pack_b_kernel(const float* __restrict__ W, __bf16* __restrict__ Wp,
                              int K, int N, int K32) {
    size_t t = (size_t)blockIdx.x * blockDim.x + threadIdx.x;
    if (t >= (size_t)K32 * N) return;
    int j = (int)(t & 15);
    int h = (int)((t >> 4) & 1);
    size_t r = t >> 5;
    int n  = (int)(r % N);
    int kt = (int)(r / N);
    int k = kt * 32 + 16 * h + j;
    Wp[t] = (k < K) ? (__bf16)W[(size_t)k * N + n] : (__bf16)0.0f;
}

// ---------------------------------------------------------------------------
// GEMM: C[M,N] = act(A @ W (+ bias)). One wave per 16x16 tile; inputs are
// pre-packed bf16 fragments, so the K-loop is branch-free straight-line code.
// ---------------------------------------------------------------------------
__global__ __launch_bounds__(32) void gemm_wmma_kernel(
    const __bf16* __restrict__ Ap, const __bf16* __restrict__ Wp,
    const float* __restrict__ bias, float* __restrict__ C,
    int M, int N, int K32, int relu)
{
    const int n0   = blockIdx.x * 16;
    const int m0   = blockIdx.y * 16;
    const int lane = threadIdx.x & 31;
    const int half = lane >> 4;
    const int l15  = lane & 15;

    const v16bf* ap = (const v16bf*)Ap;
    const v16bf* bp = (const v16bf*)Wp;
    size_t aidx = (size_t)(m0 + l15) * 2 + half;   // + kt * 2M per chunk
    size_t bidx = (size_t)(n0 + l15) * 2 + half;   // + kt * 2N per chunk
    const size_t astep = (size_t)M * 2;
    const size_t bstep = (size_t)N * 2;

    v8f acc = {0.f, 0.f, 0.f, 0.f, 0.f, 0.f, 0.f, 0.f};
    const int nkt = K32 >> 5;
    for (int kt = 0; kt < nkt; ++kt) {
        v16bf a = ap[aidx];
        v16bf b = bp[bidx];
        aidx += astep;
        bidx += bstep;
        acc = __builtin_amdgcn_wmma_f32_16x16x32_bf16(
            false, a, false, b, (short)0, acc, false, false);
    }

#pragma unroll
    for (int r = 0; r < 8; ++r) {
        int m = m0 + r + 8 * half;    // C/D: row = r + 8*half, col = lane&15
        int n = n0 + l15;
        float v = acc[r];
        if (bias) v += bias[n];
        if (relu) v = fmaxf(v, 0.0f);
        C[(size_t)m * N + n] = v;
    }
}

// ---------------------------------------------------------------------------
// Graph helpers
// ---------------------------------------------------------------------------
__global__ void zero_kernel(float* __restrict__ p, size_t n) {
    size_t t = (size_t)blockIdx.x * blockDim.x + threadIdx.x;
    if (t < n) p[t] = 0.0f;
}

__global__ void count_deg_kernel(const int* __restrict__ ei, float* __restrict__ deg, int E) {
    int t = blockIdx.x * blockDim.x + threadIdx.x;
    if (t < E) unsafeAtomicAdd(&deg[ei[E + t]], 1.0f);   // dst = ei[1][e]
}

__global__ void finalize_dinv_kernel(float* __restrict__ deg, int n) {
    int t = blockIdx.x * blockDim.x + threadIdx.x;
    if (t < n) deg[t] = rsqrtf(deg[t] + 1.0f);           // + self loop
}

// out[i,c] = h[i,c]*dinv[i]^2 + bias[c]   (self-loop term + bias)
__global__ void gcn_init_kernel(const float* __restrict__ h, const float* __restrict__ dinv,
                                const float* __restrict__ bias, float* __restrict__ out,
                                int n, int C) {
    size_t t = (size_t)blockIdx.x * blockDim.x + threadIdx.x;
    if (t >= (size_t)n * C) return;
    int i = (int)(t / C), c = (int)(t % C);
    float di = dinv[i];
    out[t] = h[t] * di * di + bias[c];
}

// out[dst,c] += h[src,c] * dinv[src]*dinv[dst]   (4 channels / thread)
__global__ void gcn_edge_kernel(const float* __restrict__ h, const int* __restrict__ ei, int E,
                                const float* __restrict__ dinv, float* __restrict__ out, int C) {
    size_t t = (size_t)blockIdx.x * blockDim.x + threadIdx.x;
    int ch4 = C >> 2;
    if (t >= (size_t)E * ch4) return;
    int e = (int)(t / ch4);
    int c = (int)(t % ch4) << 2;
    int s = ei[e], d = ei[E + e];
    float coef = dinv[s] * dinv[d];
    const float* hp = h + (size_t)s * C + c;
    float*       op = out + (size_t)d * C + c;
#pragma unroll
    for (int j = 0; j < 4; ++j) unsafeAtomicAdd(op + j, hp[j] * coef);
}

__global__ void relu_kernel(float* __restrict__ p, size_t n) {
    size_t t = (size_t)blockIdx.x * blockDim.x + threadIdx.x;
    if (t < n) p[t] = fmaxf(p[t], 0.0f);
}

__global__ void gather_rows_kernel(const float* __restrict__ src, const int* __restrict__ idx,
                                   float* __restrict__ out, int n, int C) {
    size_t t = (size_t)blockIdx.x * blockDim.x + threadIdx.x;
    if (t >= (size_t)n * C) return;
    int r = (int)(t / C), c = (int)(t % C);
    out[t] = src[(size_t)idx[r] * C + c];
}

__global__ void concat2_kernel(const float* __restrict__ a, int Ca,
                               const float* __restrict__ b, int Cb,
                               float* __restrict__ out, int n) {
    int C = Ca + Cb;
    size_t t = (size_t)blockIdx.x * blockDim.x + threadIdx.x;
    if (t >= (size_t)n * C) return;
    int r = (int)(t / C), c = (int)(t % C);
    out[t] = (c < Ca) ? a[(size_t)r * Ca + c] : b[(size_t)r * Cb + (c - Ca)];
}

__global__ void concat_bcast_kernel(const float* __restrict__ gvec, int Cg,
                                    const float* __restrict__ xb, int Cx,
                                    float* __restrict__ out, int n) {
    int C = Cg + Cx;
    size_t t = (size_t)blockIdx.x * blockDim.x + threadIdx.x;
    if (t >= (size_t)n * C) return;
    int r = (int)(t / C), c = (int)(t % C);
    out[t] = (c < Cg) ? gvec[c] : xb[(size_t)r * Cx + (c - Cg)];
}

// ---------------------------------------------------------------------------
// BatchNorm (training-mode batch stats) + ReLU, in place. One block / channel.
// ---------------------------------------------------------------------------
__global__ __launch_bounds__(256) void bnorm_relu_kernel(
    float* __restrict__ h, const float* __restrict__ gamma,
    const float* __restrict__ beta, int n, int C) {
    int c = blockIdx.x;
    __shared__ float ssum[256], ssq[256];
    float s = 0.f, sq = 0.f;
    for (int i = threadIdx.x; i < n; i += blockDim.x) {
        float v = h[(size_t)i * C + c];
        s += v; sq += v * v;
    }
    ssum[threadIdx.x] = s; ssq[threadIdx.x] = sq;
    __syncthreads();
    for (int st = 128; st > 0; st >>= 1) {
        if ((int)threadIdx.x < st) {
            ssum[threadIdx.x] += ssum[threadIdx.x + st];
            ssq[threadIdx.x]  += ssq[threadIdx.x + st];
        }
        __syncthreads();
    }
    float mu  = ssum[0] / (float)n;
    float var = ssq[0] / (float)n - mu * mu;
    float rs  = rsqrtf(var + 1e-5f) * gamma[c];
    float bb  = beta[c];
    for (int i = threadIdx.x; i < n; i += blockDim.x) {
        size_t o = (size_t)i * C + c;
        h[o] = fmaxf((h[o] - mu) * rs + bb, 0.0f);
    }
}

// column-wise max over n rows; one block per channel
__global__ __launch_bounds__(256) void colmax_kernel(
    const float* __restrict__ h, float* __restrict__ out, int n, int C) {
    int c = blockIdx.x;
    __shared__ float sm[256];
    float m = -3.4e38f;
    for (int i = threadIdx.x; i < n; i += blockDim.x)
        m = fmaxf(m, h[(size_t)i * C + c]);
    sm[threadIdx.x] = m;
    __syncthreads();
    for (int st = 128; st > 0; st >>= 1) {
        if ((int)threadIdx.x < st) sm[threadIdx.x] = fmaxf(sm[threadIdx.x], sm[threadIdx.x + st]);
        __syncthreads();
    }
    if (threadIdx.x == 0) out[c] = sm[0];
}

// ---------------------------------------------------------------------------
// KNN (k=3) inverse-squared-distance weights. One block per query point.
// ---------------------------------------------------------------------------
__global__ __launch_bounds__(128) void knn3_kernel(
    const float* __restrict__ psrc, int ns,
    const float* __restrict__ pdst, int nd,
    float* __restrict__ wout, int* __restrict__ iout) {
    __shared__ float sd[128 * 3];
    __shared__ int   si[128 * 3];
    int m = blockIdx.x;
    if (m >= nd) return;
    float q0 = pdst[m * 3], q1 = pdst[m * 3 + 1], q2 = pdst[m * 3 + 2];
    float bd[3] = {3.4e38f, 3.4e38f, 3.4e38f};
    int   bi[3] = {0, 0, 0};
    for (int j = threadIdx.x; j < ns; j += blockDim.x) {
        float a = psrc[j * 3] - q0, b = psrc[j * 3 + 1] - q1, c = psrc[j * 3 + 2] - q2;
        float d2 = a * a + b * b + c * c;
        if (d2 < bd[2]) {
            if (d2 < bd[0])      { bd[2]=bd[1]; bi[2]=bi[1]; bd[1]=bd[0]; bi[1]=bi[0]; bd[0]=d2; bi[0]=j; }
            else if (d2 < bd[1]) { bd[2]=bd[1]; bi[2]=bi[1]; bd[1]=d2;    bi[1]=j; }
            else                 { bd[2]=d2;    bi[2]=j; }
        }
    }
    int t = threadIdx.x;
#pragma unroll
    for (int r = 0; r < 3; ++r) { sd[t * 3 + r] = bd[r]; si[t * 3 + r] = bi[r]; }
    __syncthreads();
    if (t == 0) {
        float fd[3] = {3.4e38f, 3.4e38f, 3.4e38f};
        int   fi[3] = {0, 0, 0};
        for (int q = 0; q < 128 * 3; ++q) {
            float d2 = sd[q]; int j = si[q];
            if (d2 < fd[2]) {
                if (d2 < fd[0])      { fd[2]=fd[1]; fi[2]=fi[1]; fd[1]=fd[0]; fi[1]=fi[0]; fd[0]=d2; fi[0]=j; }
                else if (d2 < fd[1]) { fd[2]=fd[1]; fi[2]=fi[1]; fd[1]=d2;    fi[1]=j; }
                else                 { fd[2]=d2;    fi[2]=j; }
            }
        }
        float w0 = 1.0f / fmaxf(fd[0], 1e-16f);
        float w1 = 1.0f / fmaxf(fd[1], 1e-16f);
        float w2 = 1.0f / fmaxf(fd[2], 1e-16f);
        float ws = w0 + w1 + w2;
        wout[m * 3]     = w0 / ws;  iout[m * 3]     = fi[0];
        wout[m * 3 + 1] = w1 / ws;  iout[m * 3 + 1] = fi[1];
        wout[m * 3 + 2] = w2 / ws;  iout[m * 3 + 2] = fi[2];
    }
}

__global__ void interp3_kernel(const float* __restrict__ x, int C,
                               const int* __restrict__ kidx, const float* __restrict__ kw,
                               float* __restrict__ out, int nd) {
    size_t t = (size_t)blockIdx.x * blockDim.x + threadIdx.x;
    if (t >= (size_t)nd * C) return;
    int m = (int)(t / C), c = (int)(t % C);
    const int*   ip = kidx + m * 3;
    const float* wp = kw + m * 3;
    out[t] = wp[0] * x[(size_t)ip[0] * C + c]
           + wp[1] * x[(size_t)ip[1] * C + c]
           + wp[2] * x[(size_t)ip[2] * C + c];
}

// final tiny layer: out[m, 0..3] = h[m,:64] @ w1(64x4) + b1
__global__ void mlp_head_kernel(const float* __restrict__ h, const float* __restrict__ w1,
                                const float* __restrict__ b1, float* __restrict__ out, int n) {
    int m = blockIdx.x * blockDim.x + threadIdx.x;
    if (m >= n) return;
    float a0 = b1[0], a1 = b1[1], a2 = b1[2], a3 = b1[3];
    const float* hp = h + (size_t)m * 64;
#pragma unroll 4
    for (int c = 0; c < 64; ++c) {
        float v = hp[c];
        const float* wp = w1 + c * 4;
        a0 += v * wp[0]; a1 += v * wp[1]; a2 += v * wp[2]; a3 += v * wp[3];
    }
    float* op = out + (size_t)m * 4;
    op[0] = a0; op[1] = a1; op[2] = a2; op[3] = a3;
}

// ---------------------------------------------------------------------------
// Host-side orchestration helpers
// ---------------------------------------------------------------------------
static inline unsigned blks(size_t n) { return (unsigned)((n + 255) / 256); }

// pack A and W into bf16 WMMA fragment layouts, then run the tiled WMMA GEMM
static void gemm(const float* A, const float* W, const float* bias, float* C,
                 int M, int N, int K, int relu,
                 __bf16* Ap, __bf16* Wp, hipStream_t s) {
    int K32 = (K + 31) & ~31;
    size_t na = (size_t)M * K32;
    size_t nb = (size_t)K32 * N;
    pack_a_kernel<<<blks(na), 256, 0, s>>>(A, Ap, M, K, K32);
    pack_b_kernel<<<blks(nb), 256, 0, s>>>(W, Wp, K, N, K32);
    dim3 g((unsigned)(N / 16), (unsigned)(M / 16));
    gemm_wmma_kernel<<<g, 32, 0, s>>>(Ap, Wp, bias, C, M, N, K32, relu);
}

static void gcn_layer(const float* in, int n, int K, int N,
                      const float* W, const float* b,
                      const int* ei, int E, const float* dinv,
                      float* tmp, float* out, __bf16* Ap, __bf16* Wp, hipStream_t s) {
    gemm(in, W, nullptr, tmp, n, N, K, 0, Ap, Wp, s);              // h = x @ W
    size_t tot = (size_t)n * N;
    gcn_init_kernel<<<blks(tot), 256, 0, s>>>(tmp, dinv, b, out, n, N);
    size_t ecnt = (size_t)E * (N / 4);
    gcn_edge_kernel<<<blks(ecnt), 256, 0, s>>>(tmp, ei, E, dinv, out, N);
    relu_kernel<<<blks(tot), 256, 0, s>>>(out, tot);
}

static void degrees(const int* ei, int E, float* dinv, int n, hipStream_t s) {
    zero_kernel<<<blks(n), 256, 0, s>>>(dinv, (size_t)n);
    count_deg_kernel<<<blks(E), 256, 0, s>>>(ei, dinv, E);
    finalize_dinv_kernel<<<blks(n), 256, 0, s>>>(dinv, n);
}

extern "C" void kernel_launch(void* const* d_in, const int* in_sizes, int n_in,
                              void* d_out, int out_size, void* d_ws, size_t ws_size,
                              hipStream_t stream) {
    (void)in_sizes; (void)n_in; (void)out_size; (void)ws_size;
    const float* x    = (const float*)d_in[0];
    const float* pos  = (const float*)d_in[1];
    const float* sa1_w0=(const float*)d_in[2],  *sa1_b0=(const float*)d_in[3];
    const float* sa1_w1=(const float*)d_in[4],  *sa1_b1=(const float*)d_in[5];
    const float* sa1_w2=(const float*)d_in[6],  *sa1_b2=(const float*)d_in[7];
    const float* sa2_w0=(const float*)d_in[8],  *sa2_b0=(const float*)d_in[9];
    const float* sa2_w1=(const float*)d_in[10], *sa2_b1=(const float*)d_in[11];
    const float* sa2_w2=(const float*)d_in[12], *sa2_b2=(const float*)d_in[13];
    const float* bn_w0=(const float*)d_in[14],  *bn_b0=(const float*)d_in[15];
    const float* bn_w1=(const float*)d_in[16],  *bn_b1=(const float*)d_in[17];
    const float* bn_w2=(const float*)d_in[18],  *bn_b2=(const float*)d_in[19];
    const float* g_w0=(const float*)d_in[20],   *g_b0=(const float*)d_in[21];
    const float* g_w1=(const float*)d_in[22],   *g_b1=(const float*)d_in[23];
    const float* g_w2=(const float*)d_in[24],   *g_b2=(const float*)d_in[25];
    const float* g_g0=(const float*)d_in[26],   *g_be0=(const float*)d_in[27];
    const float* g_g1=(const float*)d_in[28],   *g_be1=(const float*)d_in[29];
    const float* fp2_w0=(const float*)d_in[30], *fp2_b0=(const float*)d_in[31];
    const float* fp2_w1=(const float*)d_in[32], *fp2_b1=(const float*)d_in[33];
    const float* fp2_w2=(const float*)d_in[34], *fp2_b2=(const float*)d_in[35];
    const float* fp1_w0=(const float*)d_in[36], *fp1_b0=(const float*)d_in[37];
    const float* fp1_w1=(const float*)d_in[38], *fp1_b1=(const float*)d_in[39];
    const float* fp1_w2=(const float*)d_in[40], *fp1_b2=(const float*)d_in[41];
    const float* m_w0=(const float*)d_in[42],   *m_b0=(const float*)d_in[43];
    const float* m_w1=(const float*)d_in[44],   *m_b1=(const float*)d_in[45];
    const int* idx0 = (const int*)d_in[46];
    const int* idx1 = (const int*)d_in[47];
    const int* ei0  = (const int*)d_in[48];
    const int* ei1  = (const int*)d_in[49];
    const int* ei2  = (const int*)d_in[50];

    // -------- workspace carve (fp32 elements; all offsets 32B aligned) -----
    float* W = (float*)d_ws;
    size_t off = 0;
    auto F = [&](size_t cnt) { float* p = W + off; off += cnt; return p; };
    float* dinv0 = F(N0_);  float* dinv1 = F(N1_);  float* dinv2 = F(N2_);
    float* A0 = F((size_t)N0_ * 256);  float* B0 = F((size_t)N0_ * 256);  float* C0 = F((size_t)N0_ * 256);
    float* x1 = F((size_t)N1_ * 64);   float* pos1 = F((size_t)N1_ * 3);  float* cat1 = F((size_t)N1_ * 67);
    float* A1 = F((size_t)N1_ * 1024); float* B1 = F((size_t)N1_ * 1024); float* C1 = F((size_t)N1_ * 1024);
    float* x2 = F((size_t)N2_ * 128);  float* pos2 = F((size_t)N2_ * 3);  float* cat2 = F((size_t)N2_ * 131);
    float* A2 = F((size_t)N2_ * 256);  float* B2 = F((size_t)N2_ * 256);  float* C2 = F((size_t)N2_ * 256);
    float* g0 = F((size_t)N2_ * 512);  float* g1 = F((size_t)N2_ * 1024); float* g2 = F((size_t)N2_ * 2048);
    float* gmax = F(2048);
    float* hg = F((size_t)N2_ * 2304);
    float* kw2 = F((size_t)N1_ * 3);   int* ki2 = (int*)F((size_t)N1_ * 3);
    float* it2 = F((size_t)N1_ * 2304);
    float* catf2 = F((size_t)N1_ * 2368);
    float* kw1 = F((size_t)N0_ * 3);   int* ki1 = (int*)F((size_t)N0_ * 3);
    float* it1 = F((size_t)N0_ * 512);
    float* catf1 = F((size_t)N0_ * 518);
    // bf16 fragment scratch: max A = 4096x2368 bf16, max W = 2368x1024 bf16
    __bf16* Ap = (__bf16*)F(4900000);   // >= 9,699,328 bf16
    __bf16* Wp = (__bf16*)F(1250000);   // >= 2,424,832 bf16

    hipStream_t s = stream;

    // -------- normalized degrees per level --------
    degrees(ei0, E0_, dinv0, N0_, s);
    degrees(ei1, E1_, dinv1, N1_, s);
    degrees(ei2, E2_, dinv2, N2_, s);

    // -------- SA1 (level-0 graph) --------
    gcn_layer(x,  N0_, 6,  32, sa1_w0, sa1_b0, ei0, E0_, dinv0, A0, B0, Ap, Wp, s);
    gcn_layer(B0, N0_, 32, 32, sa1_w1, sa1_b1, ei0, E0_, dinv0, A0, C0, Ap, Wp, s);
    gcn_layer(C0, N0_, 32, 64, sa1_w2, sa1_b2, ei0, E0_, dinv0, A0, B0, Ap, Wp, s);
    gather_rows_kernel<<<blks((size_t)N1_ * 64), 256, 0, s>>>(B0, idx0, x1, N1_, 64);
    gather_rows_kernel<<<blks((size_t)N1_ * 3),  256, 0, s>>>(pos, idx0, pos1, N1_, 3);

    // -------- SA2 (level-1 graph) --------
    concat2_kernel<<<blks((size_t)N1_ * 67), 256, 0, s>>>(x1, 64, pos1, 3, cat1, N1_);
    gcn_layer(cat1, N1_, 67, 64,  sa2_w0, sa2_b0, ei1, E1_, dinv1, A1, B1, Ap, Wp, s);
    gcn_layer(B1,   N1_, 64, 64,  sa2_w1, sa2_b1, ei1, E1_, dinv1, A1, C1, Ap, Wp, s);
    gcn_layer(C1,   N1_, 64, 128, sa2_w2, sa2_b2, ei1, E1_, dinv1, A1, B1, Ap, Wp, s);
    gather_rows_kernel<<<blks((size_t)N2_ * 128), 256, 0, s>>>(B1, idx1, x2, N2_, 128);
    gather_rows_kernel<<<blks((size_t)N2_ * 3),   256, 0, s>>>(pos1, idx1, pos2, N2_, 3);

    // -------- bottleneck (level-2 graph) --------
    concat2_kernel<<<blks((size_t)N2_ * 131), 256, 0, s>>>(x2, 128, pos2, 3, cat2, N2_);
    gcn_layer(cat2, N2_, 131, 128, bn_w0, bn_b0, ei2, E2_, dinv2, A2, B2, Ap, Wp, s);
    gcn_layer(B2,   N2_, 128, 128, bn_w1, bn_b1, ei2, E2_, dinv2, A2, C2, Ap, Wp, s);
    gcn_layer(C2,   N2_, 128, 256, bn_w2, bn_b2, ei2, E2_, dinv2, A2, B2, Ap, Wp, s); // xb = B2

    // -------- GlobalSA: MLP + BN + max-pool --------
    gemm(cat2, g_w0, g_b0, g0, N2_, 512, 131, 0, Ap, Wp, s);
    bnorm_relu_kernel<<<512, 256, 0, s>>>(g0, g_g0, g_be0, N2_, 512);
    gemm(g0, g_w1, g_b1, g1, N2_, 1024, 512, 0, Ap, Wp, s);
    bnorm_relu_kernel<<<1024, 256, 0, s>>>(g1, g_g1, g_be1, N2_, 1024);
    gemm(g1, g_w2, g_b2, g2, N2_, 2048, 1024, 0, Ap, Wp, s);
    colmax_kernel<<<2048, 256, 0, s>>>(g2, gmax, N2_, 2048);
    concat_bcast_kernel<<<blks((size_t)N2_ * 2304), 256, 0, s>>>(gmax, 2048, B2, 256, hg, N2_);

    // -------- FP2: interpolate level2 -> level1, GCNs on level-1 graph -----
    knn3_kernel<<<N1_, 128, 0, s>>>(pos2, N2_, pos1, N1_, kw2, ki2);
    interp3_kernel<<<blks((size_t)N1_ * 2304), 256, 0, s>>>(hg, 2304, ki2, kw2, it2, N1_);
    concat2_kernel<<<blks((size_t)N1_ * 2368), 256, 0, s>>>(it2, 2304, x1, 64, catf2, N1_);
    gcn_layer(catf2, N1_, 2368, 1024, fp2_w0, fp2_b0, ei1, E1_, dinv1, A1, B1, Ap, Wp, s);
    gcn_layer(B1,    N1_, 1024, 1024, fp2_w1, fp2_b1, ei1, E1_, dinv1, A1, C1, Ap, Wp, s);
    gcn_layer(C1,    N1_, 1024, 512,  fp2_w2, fp2_b2, ei1, E1_, dinv1, A1, B1, Ap, Wp, s);

    // -------- FP1: interpolate level1 -> level0, GCNs on level-0 graph -----
    knn3_kernel<<<N0_, 128, 0, s>>>(pos1, N1_, pos, N0_, kw1, ki1);
    interp3_kernel<<<blks((size_t)N0_ * 512), 256, 0, s>>>(B1, 512, ki1, kw1, it1, N0_);
    concat2_kernel<<<blks((size_t)N0_ * 518), 256, 0, s>>>(it1, 512, x, 6, catf1, N0_);
    gcn_layer(catf1, N0_, 518, 256, fp1_w0, fp1_b0, ei0, E0_, dinv0, A0, B0, Ap, Wp, s);
    gcn_layer(B0,    N0_, 256, 256, fp1_w1, fp1_b1, ei0, E0_, dinv0, A0, C0, Ap, Wp, s);
    gcn_layer(C0,    N0_, 256, 128, fp1_w2, fp1_b2, ei0, E0_, dinv0, A0, B0, Ap, Wp, s);

    // -------- head MLP --------
    gemm(B0, m_w0, m_b0, C0, N0_, 64, 128, 1, Ap, Wp, s);
    mlp_head_kernel<<<blks(N0_), 256, 0, s>>>(C0, m_w1, m_b1, (float*)d_out, N0_);
}
